// ProductQuantizer_88244398063982
// MI455X (gfx1250) — compile-verified, matrix operand order checked by reading, not run on hardware
//
#include <hip/hip_runtime.h>
#include <hip/hip_bf16.h>

typedef __attribute__((ext_vector_type(16))) __bf16 v16bf;
typedef __attribute__((ext_vector_type(8)))  __bf16 v8bf;
typedef __attribute__((ext_vector_type(8)))  float  v8f;

#define B_   32
#define S_   4096
#define F_   256
#define G_   2
#define N_   128
#define D_   128
#define E_   256
#define TOK_ (B_ * S_)

// ---------------------------------------------------------------------------
// Prep: convert W_logits [G,D,N] and W_out [F,E] (f32, row-major) into bf16
// WMMA B-fragments (16x16x32). Fragment = 32 lanes x 16 bf16, contiguous, so
// the main kernel loads a lane's operand with one 32B coalesced access.
// Per ISA 7.12.2 (16-bit operands): lane -> column n = lane&15,
// element e -> k = e + 8*(e>=8) + 8*(lane>=16).
// ---------------------------------------------------------------------------
__global__ void pq_prep_weights(const float* __restrict__ Wl,
                                const float* __restrict__ Wout,
                                __bf16* __restrict__ wlFrag,
                                __bf16* __restrict__ woutFrag) {
  int tid    = blockIdx.x * 256 + threadIdx.x;   // 98304 total
  int fragG  = tid >> 9;                         // fragment id
  int within = tid & 511;                        // 32 lanes * 16 elems
  int lane   = within >> 4;
  int e      = within & 15;
  int kf     = e + (e & 8) + ((lane & 16) >> 1); // k within 32-wide K step
  int n16    = lane & 15;

  if (fragG < 64) {                              // W_logits: 2 groups * 8 ntiles * 4 ksteps
    int g   = fragG >> 5;
    int rem = fragG & 31;
    int nt  = rem >> 2;
    int kt  = rem & 3;
    int k   = kt * 32 + kf;
    int n   = nt * 16 + n16;
    wlFrag[((size_t)fragG << 9) + within] = (__bf16)Wl[(((g << 7) + k) << 7) + n];
  } else {                                       // W_out: 16 ntiles * 8 ksteps
    int f2 = fragG - 64;
    int nt = f2 >> 3;
    int kt = f2 & 7;
    int k  = kt * 32 + kf;
    int n  = nt * 16 + n16;
    woutFrag[((size_t)f2 << 9) + within] = (__bf16)Wout[(k << 8) + n];
  }
}

// ---------------------------------------------------------------------------
// Fused PQ kernel: one block = 16 tokens, 8 wave32.
// logits(WMMA bf16) -> +gumbel+bias -> argmax (softmax skipped: ST forward
// value == hard one-hot) -> codebook row gather -> out GEMM (WMMA bf16).
// ---------------------------------------------------------------------------
__launch_bounds__(256, 2)
__global__ void pq_fused(const float* __restrict__ features,
                         const float* __restrict__ gumbel,
                         const float* __restrict__ b_logits,
                         const float* __restrict__ codebooks,
                         const float* __restrict__ b_out,
                         const __bf16* __restrict__ wlFrag,
                         const __bf16* __restrict__ woutFrag,
                         float* __restrict__ out,
                         int* __restrict__ indices) {
  constexpr int LPB = 264;   // bf16 LDS row pitch (multiple of 8 -> 16B aligned runs)
  constexpr int LPF = 260;   // f32 LDS row pitch
  __shared__ __bf16 sFeat[16 * LPB];
  __shared__ float  sLogits[16 * LPF];
  __shared__ __bf16 sQuant[16 * LPB];
  __shared__ int    sIdx[32];

  const int t       = threadIdx.x;
  const int lane    = t & 31;
  const int wave    = t >> 5;
  const int tokBase = blockIdx.x << 4;
  const int m       = lane & 15;
  const int hi      = (lane & 16) >> 1;   // 0 or 8 (K / row offset for upper half-wave)

  // ---- Phase 0: stage feature tile [16,256] f32 -> bf16 LDS (coalesced) ----
  const float* fsrc = features + ((size_t)tokBase << 8);
#pragma unroll
  for (int i = 0; i < 4; ++i) {
    int off = i * 1024 + t * 4;
    float4 v = *(const float4*)(fsrc + off);
    int r = off >> 8, c = off & 255;
    __bf16* d = &sFeat[r * LPB + c];
    d[0] = (__bf16)v.x; d[1] = (__bf16)v.y; d[2] = (__bf16)v.z; d[3] = (__bf16)v.w;
  }
  __syncthreads();

  // ---- Phase 1: logits = feat @ W_logits  (16 x 256, K=128 per group) ----
#pragma unroll
  for (int tt = 0; tt < 2; ++tt) {
    int nt = wave * 2 + tt;        // 0..15 output column tile
    int cn = nt << 4;
    int g  = cn >> 7;
    int nk = (cn & 127) >> 4;      // n-tile within group
    v8f acc = {};
#pragma unroll
    for (int kt = 0; kt < 4; ++kt) {
      const __bf16* arow = &sFeat[m * LPB + (g << 7) + kt * 32 + hi];
      v8bf lo = *(const v8bf*)(arow);
      v8bf hh = *(const v8bf*)(arow + 16);
      v16bf a;
#pragma unroll
      for (int e2 = 0; e2 < 8; ++e2) { a[e2] = lo[e2]; a[e2 + 8] = hh[e2]; }
      int frag = (((g << 3) | nk) << 2) | kt;
      v16bf b = *(const v16bf*)(wlFrag + ((size_t)frag << 9) + (lane << 4));
      acc = __builtin_amdgcn_wmma_f32_16x16x32_bf16(false, a, false, b,
                                                    (short)0, acc, false, false);
    }
#pragma unroll
    for (int r = 0; r < 8; ++r)
      sLogits[(r + hi) * LPF + cn + m] = acc[r];
  }
  __syncthreads();

  // ---- Phase 2: argmax over (logits + b_logits + gumbel). Softmax skipped:
  // st = (hard - soft) + soft == one-hot in forward value. Tie-break -> lowest
  // index to match jnp.argmax first-max semantics. ----
#pragma unroll
  for (int i = 0; i < 4; ++i) {
    int p  = wave * 4 + i;         // 0..31 = (token row, group)
    int pm = p >> 1, pg = p & 1;
    int token = tokBase + pm;
    float best = -3.402823466e38f;
    int bidx = 0;
#pragma unroll
    for (int j = 0; j < 4; ++j) {
      int n = j * 32 + lane;
      int c = (pg << 7) + n;
      float v = sLogits[pm * LPF + c] + b_logits[c] + gumbel[((size_t)token << 8) + c];
      if (v > best) { best = v; bidx = n; }
    }
#pragma unroll
    for (int offx = 16; offx > 0; offx >>= 1) {
      float ov = __shfl_xor(best, offx, 32);
      int   oi = __shfl_xor(bidx, offx, 32);
      if (ov > best || (ov == best && oi < bidx)) { best = ov; bidx = oi; }
    }
    if (lane == 0) {
      sIdx[p] = bidx;
      indices[((size_t)token << 1) + pg] = bidx;
    }
  }
  __syncthreads();

  // ---- Phase 3: quantized = codebooks[g, argmax, :]  (gather, f32 -> bf16) ----
  {
    int p   = t >> 3;              // 0..31 (row, group) pairs, 8 threads each
    int sub = t & 7;
    int pm  = p >> 1, pg = p & 1;
    int idx = sIdx[p];
    const float* crow = codebooks + (((size_t)((pg << 7) + idx)) << 7);
#pragma unroll
    for (int j = 0; j < 4; ++j) {
      int c0 = sub * 16 + j * 4;
      float4 v = *(const float4*)(crow + c0);
      __bf16* d = &sQuant[pm * LPB + (pg << 7) + c0];
      d[0] = (__bf16)v.x; d[1] = (__bf16)v.y; d[2] = (__bf16)v.z; d[3] = (__bf16)v.w;
    }
  }
  __syncthreads();

  // ---- Phase 4: out = quantized @ W_out + b_out  (16 x 256, K=256) ----
#pragma unroll
  for (int tt = 0; tt < 2; ++tt) {
    int nt = wave * 2 + tt;        // 0..15
    int cn = nt << 4;
    v8f acc = {};
#pragma unroll
    for (int kt = 0; kt < 8; ++kt) {
      const __bf16* arow = &sQuant[m * LPB + kt * 32 + hi];
      v8bf lo = *(const v8bf*)(arow);
      v8bf hh = *(const v8bf*)(arow + 16);
      v16bf a;
#pragma unroll
      for (int e2 = 0; e2 < 8; ++e2) { a[e2] = lo[e2]; a[e2 + 8] = hh[e2]; }
      int frag = (nt << 3) + kt;
      v16bf b = *(const v16bf*)(woutFrag + ((size_t)frag << 9) + (lane << 4));
      acc = __builtin_amdgcn_wmma_f32_16x16x32_bf16(false, a, false, b,
                                                    (short)0, acc, false, false);
    }
    int col = cn + m;
    float bias = b_out[col];
#pragma unroll
    for (int r = 0; r < 8; ++r)
      out[((size_t)(tokBase + r + hi) << 8) + col] = acc[r] + bias;
  }
}

extern "C" void kernel_launch(void* const* d_in, const int* in_sizes, int n_in,
                              void* d_out, int out_size, void* d_ws, size_t ws_size,
                              hipStream_t stream) {
  const float* features  = (const float*)d_in[0];
  const float* gumbel    = (const float*)d_in[1];
  const float* W_logits  = (const float*)d_in[2];
  const float* b_logits  = (const float*)d_in[3];
  const float* codebooks = (const float*)d_in[4];
  const float* W_out     = (const float*)d_in[5];
  const float* b_out     = (const float*)d_in[6];

  __bf16* wlFrag   = (__bf16*)d_ws;              // 64 frags * 512 bf16 = 64 KB
  __bf16* woutFrag = wlFrag + 64 * 512;          // 128 frags * 512 bf16 = 128 KB
  float*  out      = (float*)d_out;
  int*    indices  = (int*)d_out + (size_t)TOK_ * E_;

  pq_prep_weights<<<384, 256, 0, stream>>>(W_logits, W_out, wlFrag, woutFrag);
  pq_fused<<<TOK_ / 16, 256, 0, stream>>>(features, gumbel, b_logits, codebooks,
                                          b_out, wlFrag, woutFrag, out, indices);
}
